// CausalSelfAttention_5171140624929
// MI455X (gfx1250) — matrix operand order repaired and checked by reading, not compiled, not run on hardware
//
#include <hip/hip_runtime.h>
#include <hip/hip_bf16.h>

// CDNA5 / gfx1250 flash-attention pipeline using v_wmma_f32_16x16x32_f16.
// B=2, L=2048, H=16, D=64, E=1024.

typedef __attribute__((ext_vector_type(16))) _Float16 v16h;
typedef __attribute__((ext_vector_type(8)))  float    v8f;

#define BB 2
#define LL 2048
#define HH 16
#define DD 64
#define EE 1024

__device__ __forceinline__ v8f wmma16(v16h a, v16h b, v8f c) {
  // 8 args: (neg_a, A, neg_b, B, c_mod, C, reuse_a, reuse_b)
  return __builtin_amdgcn_wmma_f32_16x16x32_f16(false, a, false, b, (short)0, c, false, false);
}

// A-operand (16x32 f16, M = lane&15): per the ISA table, lane holds two
// contiguous 8-half chunks at K = (lane>>4)*8 and K = 16 + (lane>>4)*8.
__device__ __forceinline__ v16h load_a_tile(const _Float16* p, int ld, int row,
                                            int k0, int lane) {
  const _Float16* base = p + (size_t)row * ld + k0 + ((lane >> 4) & 1) * 8;
  v16h a;
  ((uint4*)&a)[0] = *(const uint4*)(base);
  ((uint4*)&a)[1] = *(const uint4*)(base + 16);
  return a;
}

// B-operand (32x16 f16, N = lane&15): lane holds two contiguous 8-half chunks
// at K = (lane>>4)*8 and K = 16 + (lane>>4)*8 (same order as A).
__device__ __forceinline__ v16h load_b_tile(const _Float16* p, int ld, int nrow,
                                            int k0, int lane) {
  const _Float16* base = p + (size_t)nrow * ld + k0 + ((lane >> 4) & 1) * 8;
  v16h b;
  ((uint4*)&b)[0] = *(const uint4*)(base);
  ((uint4*)&b)[1] = *(const uint4*)(base + 16);
  return b;
}

__device__ __forceinline__ float half_max16(float v) {
  #pragma unroll
  for (int m = 1; m < 16; m <<= 1) v = fmaxf(v, __shfl_xor(v, m, 32));
  return v;
}
__device__ __forceinline__ float half_sum16(float v) {
  #pragma unroll
  for (int m = 1; m < 16; m <<= 1) v += __shfl_xor(v, m, 32);
  return v;
}

// ---------------------------------------------------------------------------
// Kernel 0: convert f32 inputs to f16 staging buffers.
// ---------------------------------------------------------------------------
#define NX (BB * LL * EE)        // 4194304
#define NWQ (3 * EE * EE)        // 3145728
#define NWO (EE * EE)            // 1048576

__global__ void cvt_kernel(const float* __restrict__ x,
                           const float* __restrict__ wq,
                           const float* __restrict__ wo,
                           _Float16* __restrict__ xh,
                           _Float16* __restrict__ wqh,
                           _Float16* __restrict__ woh) {
  int i = blockIdx.x * blockDim.x + threadIdx.x;
  if (i < NX) xh[i] = (_Float16)x[i];
  else if (i < NX + NWQ) wqh[i - NX] = (_Float16)wq[i - NX];
  else if (i < NX + NWQ + NWO) woh[i - NX - NWQ] = (_Float16)wo[i - NX - NWQ];
}

// ---------------------------------------------------------------------------
// Kernel 1: QKV projection. out[m,n] = sum_k X[m,k]*Wqkv[n,k].
// M=4096, N=3072, K=1024. Per-wave 32x64 tile (2x4 C tiles).
// Epilogue scatters: Q scaled by 1/8 -> (B,H,L,D); K -> (B,H,L,D);
// V transposed -> (B,H,D,L).
// ---------------------------------------------------------------------------
__global__ void qkv_gemm_kernel(const _Float16* __restrict__ Xh,
                                const _Float16* __restrict__ Wh,
                                _Float16* __restrict__ Qh,
                                _Float16* __restrict__ Kh,
                                _Float16* __restrict__ Vt) {
  const int lane = threadIdx.x & 31;
  const int wid = (blockIdx.x * blockDim.x + threadIdx.x) >> 5;
  const int NT = (3 * EE) / 64;  // 48
  const int m0 = (wid / NT) * 32;
  const int n0 = (wid % NT) * 64;

  v8f acc[2][4];
  #pragma unroll
  for (int mi = 0; mi < 2; ++mi)
    #pragma unroll
    for (int ni = 0; ni < 4; ++ni) acc[mi][ni] = (v8f){};

  for (int k0 = 0; k0 < EE; k0 += 32) {
    v16h a0 = load_a_tile(Xh, EE, m0 + (lane & 15), k0, lane);
    v16h a1 = load_a_tile(Xh, EE, m0 + 16 + (lane & 15), k0, lane);
    #pragma unroll
    for (int ni = 0; ni < 4; ++ni) {
      v16h bt = load_b_tile(Wh, EE, n0 + ni * 16 + (lane & 15), k0, lane);
      acc[0][ni] = wmma16(a0, bt, acc[0][ni]);
      acc[1][ni] = wmma16(a1, bt, acc[1][ni]);
    }
  }

  const int hi8 = (lane >> 4) * 8;
  #pragma unroll
  for (int mi = 0; mi < 2; ++mi)
    #pragma unroll
    for (int ni = 0; ni < 4; ++ni)
      #pragma unroll
      for (int i = 0; i < 8; ++i) {
        const int m = m0 + mi * 16 + i + hi8;       // b*L + l
        const int n = n0 + ni * 16 + (lane & 15);   // qkv channel
        const float v = acc[mi][ni][i];
        const int b = m >> 11, lq = m & (LL - 1);
        const int which = n >> 10, c = n & (EE - 1);
        const int h = c >> 6, d = c & 63;
        const size_t bh = (size_t)(b * HH + h);
        if (which == 0)
          Qh[(bh * LL + lq) * DD + d] = (_Float16)(v * 0.125f);  // 1/sqrt(64)
        else if (which == 1)
          Kh[(bh * LL + lq) * DD + d] = (_Float16)v;
        else
          Vt[(bh * DD + d) * LL + lq] = (_Float16)v;             // transposed
      }
}

// ---------------------------------------------------------------------------
// Kernel 2: causal flash attention. One wave per (b,h,16-row q tile).
// Online softmax; P goes through LDS to re-layout C->A operand.
// ---------------------------------------------------------------------------
__global__ void flash_kernel(const _Float16* __restrict__ Qh,
                             const _Float16* __restrict__ Kh,
                             const _Float16* __restrict__ Vt,
                             _Float16* __restrict__ AOh) {
  __shared__ _Float16 smem[8][16 * 40];  // per-wave 16x32 P tile, padded ld=40
  const int lane = threadIdx.x & 31;
  const int w = threadIdx.x >> 5;
  const int gw = blockIdx.x * 8 + w;
  const int bh = gw >> 7;          // 128 q-tiles per (b,h)
  const int q0 = (gw & 127) * 16;
  const int b = bh >> 4, h = bh & 15;
  const int hi8 = (lane >> 4) * 8;

  const _Float16* Qb = Qh + (size_t)bh * LL * DD;
  const _Float16* Kb = Kh + (size_t)bh * LL * DD;
  const _Float16* Vb = Vt + (size_t)bh * DD * LL;

  // Q tile (pre-scaled), D=64 -> two K=32 chunks held in registers.
  const v16h qa0 = load_a_tile(Qb, DD, q0 + (lane & 15), 0, lane);
  const v16h qa1 = load_a_tile(Qb, DD, q0 + (lane & 15), 32, lane);

  v8f o[4];
  #pragma unroll
  for (int t = 0; t < 4; ++t) o[t] = (v8f){};
  float rm[8], rs[8];
  #pragma unroll
  for (int i = 0; i < 8; ++i) { rm[i] = -1e30f; rs[i] = 0.0f; }

  const int nb = (q0 + 16 + 31) >> 5;  // 32-key blocks covering keys <= q0+15
  for (int kb = 0; kb < nb; ++kb) {
    const int key0 = kb * 32;
    // S = Q @ K^T : two 16-wide key tiles.
    v8f s0 = (v8f){}, s1 = (v8f){};
    {
      v16h bt = load_b_tile(Kb, DD, key0 + (lane & 15), 0, lane);
      s0 = wmma16(qa0, bt, s0);
      bt = load_b_tile(Kb, DD, key0 + (lane & 15), 32, lane);
      s0 = wmma16(qa1, bt, s0);
      bt = load_b_tile(Kb, DD, key0 + 16 + (lane & 15), 0, lane);
      s1 = wmma16(qa0, bt, s1);
      bt = load_b_tile(Kb, DD, key0 + 16 + (lane & 15), 32, lane);
      s1 = wmma16(qa1, bt, s1);
    }
    // Causal mask on diagonal-straddling blocks.
    if (key0 + 31 > q0) {
      #pragma unroll
      for (int i = 0; i < 8; ++i) {
        const int row = q0 + i + hi8;
        const int c0 = key0 + (lane & 15);
        if (c0 > row)      s0[i] = -1e30f;
        if (c0 + 16 > row) s1[i] = -1e30f;
      }
    }
    // Online softmax update (row = 16 lanes within each lane-half).
    float fs[8];
    #pragma unroll
    for (int i = 0; i < 8; ++i) {
      float t = half_max16(fmaxf(s0[i], s1[i]));
      const float mn = fmaxf(rm[i], t);
      fs[i] = __expf(rm[i] - mn);
      rm[i] = mn;
      const float p0 = __expf(s0[i] - mn);
      const float p1 = __expf(s1[i] - mn);
      s0[i] = p0; s1[i] = p1;
      rs[i] = rs[i] * fs[i] + half_sum16(p0 + p1);
    }
    #pragma unroll
    for (int t = 0; t < 4; ++t)
      #pragma unroll
      for (int i = 0; i < 8; ++i) o[t][i] *= fs[i];

    // P (f32 C-layout) -> LDS f16 row-major -> reload in A-operand layout.
    #pragma unroll
    for (int i = 0; i < 8; ++i) {
      const int row = i + hi8;
      smem[w][row * 40 + (lane & 15)]      = (_Float16)s0[i];
      smem[w][row * 40 + 16 + (lane & 15)] = (_Float16)s1[i];
    }
    asm volatile("s_wait_dscnt 0" ::: "memory");
    const v16h pa = load_a_tile(&smem[w][0], 40, lane & 15, 0, lane);

    // O += P @ V : V^T stored (D,L), so B-operand rows are contiguous in key.
    #pragma unroll
    for (int t = 0; t < 4; ++t) {
      v16h vb = load_b_tile(Vb, LL, t * 16 + (lane & 15), key0, lane);
      o[t] = wmma16(pa, vb, o[t]);
    }
    asm volatile("" ::: "memory");  // keep next iter's DS stores after loads
  }

  // Normalize and store attention output as f16 (B, L, H*D) = (4096, 1024).
  #pragma unroll
  for (int t = 0; t < 4; ++t)
    #pragma unroll
    for (int i = 0; i < 8; ++i) {
      const int row = q0 + i + hi8;
      const float val = o[t][i] / rs[i];
      AOh[((size_t)(b * LL + row)) * EE + h * DD + t * 16 + (lane & 15)] =
          (_Float16)val;
    }
}

// ---------------------------------------------------------------------------
// Kernel 3: output projection. y[m,n] = sum_k AO[m,k]*Wout[n,k], f32 out.
// M=4096, N=1024, K=1024; per-wave 32x64 tile.
// ---------------------------------------------------------------------------
__global__ void out_gemm_kernel(const _Float16* __restrict__ AOh,
                                const _Float16* __restrict__ Woh,
                                float* __restrict__ y) {
  const int lane = threadIdx.x & 31;
  const int wid = (blockIdx.x * blockDim.x + threadIdx.x) >> 5;
  const int NT = EE / 64;  // 16
  const int m0 = (wid / NT) * 32;
  const int n0 = (wid % NT) * 64;

  v8f acc[2][4];
  #pragma unroll
  for (int mi = 0; mi < 2; ++mi)
    #pragma unroll
    for (int ni = 0; ni < 4; ++ni) acc[mi][ni] = (v8f){};

  for (int k0 = 0; k0 < EE; k0 += 32) {
    v16h a0 = load_a_tile(AOh, EE, m0 + (lane & 15), k0, lane);
    v16h a1 = load_a_tile(AOh, EE, m0 + 16 + (lane & 15), k0, lane);
    #pragma unroll
    for (int ni = 0; ni < 4; ++ni) {
      v16h bt = load_b_tile(Woh, EE, n0 + ni * 16 + (lane & 15), k0, lane);
      acc[0][ni] = wmma16(a0, bt, acc[0][ni]);
      acc[1][ni] = wmma16(a1, bt, acc[1][ni]);
    }
  }

  const int hi8 = (lane >> 4) * 8;
  #pragma unroll
  for (int mi = 0; mi < 2; ++mi)
    #pragma unroll
    for (int ni = 0; ni < 4; ++ni)
      #pragma unroll
      for (int i = 0; i < 8; ++i) {
        const int m = m0 + mi * 16 + i + hi8;
        const int n = n0 + ni * 16 + (lane & 15);
        y[(size_t)m * EE + n] = acc[mi][ni][i];
      }
}

// ---------------------------------------------------------------------------
extern "C" void kernel_launch(void* const* d_in, const int* in_sizes, int n_in,
                              void* d_out, int out_size, void* d_ws, size_t ws_size,
                              hipStream_t stream) {
  const float* net_in = (const float*)d_in[0];  // (B,L,E) f32
  const float* W_qkv  = (const float*)d_in[1];  // (3E,E) f32
  const float* W_out  = (const float*)d_in[2];  // (E,E)  f32
  float* out = (float*)d_out;                   // (B,L,E) f32

  // Workspace layout (f16), 48 MB total.
  _Float16* Xh    = (_Float16*)d_ws;            // 4096x1024
  _Float16* Wqkvh = Xh + (size_t)NX;            // 3072x1024
  _Float16* Wouth = Wqkvh + (size_t)NWQ;        // 1024x1024
  _Float16* Qh    = Wouth + (size_t)NWO;        // (B,H,L,D)
  _Float16* Kh    = Qh + (size_t)NX;            // (B,H,L,D)
  _Float16* Vt    = Kh + (size_t)NX;            // (B,H,D,L)
  _Float16* AOh   = Vt + (size_t)NX;            // 4096x1024

  const int cvt_total = NX + NWQ + NWO;
  cvt_kernel<<<(cvt_total + 255) / 256, 256, 0, stream>>>(
      net_in, W_qkv, W_out, Xh, Wqkvh, Wouth);

  // 128 M-tiles x 48 N-tiles = 6144 waves = 768 blocks of 8 waves.
  qkv_gemm_kernel<<<768, 256, 0, stream>>>(Xh, Wqkvh, Qh, Kh, Vt);

  // B*H*(L/16) = 4096 waves = 512 blocks.
  flash_kernel<<<512, 256, 0, stream>>>(Qh, Kh, Vt, AOh);

  // 128 M-tiles x 16 N-tiles = 2048 waves = 256 blocks.
  out_gemm_kernel<<<256, 256, 0, stream>>>(AOh, Wouth, out);
}
